// LinearAttention_40870908789314
// MI455X (gfx1250) — matrix operand order repaired and checked by reading, not compiled, not run on hardware
//
#include <hip/hip_runtime.h>

typedef __attribute__((ext_vector_type(16))) __bf16 v16bf;
typedef __attribute__((ext_vector_type(8)))  __bf16 bf16x8;
typedef __attribute__((ext_vector_type(4)))  __bf16 bf16x4;
typedef __attribute__((ext_vector_type(8)))  float  v8f;

#define BATCH 16
#define CD    256   // DIM
#define HD    512   // HIDDEN
#define NN    4096  // 64*64
#define NHEAD 8
#define HC    64

// ---- WMMA helper (layouts per CDNA5 ISA 7.12.2, wave32) --------------------
__device__ __forceinline__ v8f wmma_bf16(v16bf a, v16bf b, v8f c) {
  return __builtin_amdgcn_wmma_f32_16x16x32_bf16(false, a, false, b, (short)0, c,
                                                 false, false);
}

// A fragment from a row-major [M][K] bf16 source: per-lane data is two
// contiguous 16B runs:  a[i] = src[row][8*half + i], a[8+i] = src[row][16+8*half+i]
__device__ __forceinline__ v16bf frag_a(const __bf16* row_ptr, int half) {
  const bf16x8 lo = *(const bf16x8*)(row_ptr + 8 * half);
  const bf16x8 hi = *(const bf16x8*)(row_ptr + 16 + 8 * half);
  v16bf a;
#pragma unroll
  for (int i = 0; i < 8; ++i) { a[i] = lo[i]; a[8 + i] = hi[i]; }
  return a;
}

// B fragment from K-contiguous-per-column storage: b[i] = col[8*half + i], b[8+i] = col[16+8*half+i]
__device__ __forceinline__ v16bf frag_b(const __bf16* col_ptr, int half) {
  const bf16x8 b0 = *(const bf16x8*)(col_ptr + 8 * half);
  const bf16x8 b1 = *(const bf16x8*)(col_ptr + 16 + 8 * half);
  v16bf b;
#pragma unroll
  for (int i = 0; i < 8; ++i) { b[i] = b0[i]; b[8 + i] = b1[i]; }
  return b;
}

// Stage a 128x32 f32 A-tile (row-major) into LDS as bf16, float4-vectorized.
__device__ __forceinline__ void stage_a_f32(__bf16* As, const float* src,
                                            size_t ld, int tid) {
#pragma unroll
  for (int i = 0; i < 4; ++i) {
    const int g = tid + i * 256;        // quad index
    const int r = g >> 3;               // 0..127
    const int c = (g & 7) * 4;          // 0..28
    const float4 f = *(const float4*)(src + (size_t)r * ld + c);
    bf16x4 pk = {(__bf16)f.x, (__bf16)f.y, (__bf16)f.z, (__bf16)f.w};
    *(bf16x4*)&As[r * 32 + c] = pk;     // ds_store_b64
  }
}

// Stage a 32x128 f32 B-tile into LDS *transposed* (Bt[n][k]), float4 loads.
__device__ __forceinline__ void stage_bt_f32(__bf16* Bt, const float* src,
                                             size_t ld, int tid) {
#pragma unroll
  for (int i = 0; i < 4; ++i) {
    const int g = tid + i * 256;
    const int r = g >> 5;               // k row 0..31
    const int c = (g & 31) * 4;         // n col 0..124
    const float4 f = *(const float4*)(src + (size_t)r * ld + c);
    Bt[(c + 0) * 32 + r] = (__bf16)f.x;
    Bt[(c + 1) * 32 + r] = (__bf16)f.y;
    Bt[(c + 2) * 32 + r] = (__bf16)f.z;
    Bt[(c + 3) * 32 + r] = (__bf16)f.w;
  }
}

// Same but bf16 source (16B loads).
__device__ __forceinline__ void stage_bt_bf16(__bf16* Bt, const __bf16* src,
                                              size_t ld, int tid) {
#pragma unroll
  for (int i = 0; i < 2; ++i) {
    const int g = tid + i * 256;
    const int r = g >> 4;               // k row 0..31
    const int c = (g & 15) * 8;         // n col 0..120
    const bf16x8 f = *(const bf16x8*)(src + (size_t)r * ld + c);
#pragma unroll
    for (int t = 0; t < 8; ++t) Bt[(c + t) * 32 + r] = f[t];
  }
}

// ---- Kernel 1: fused QKV projection  O[sel] = W[sel] @ x[b]  (bf16 out) ----
// grid (NN/128, 512/128, 16*3), block 256 (8 waves). Each wave: 16M x 128N.
__global__ void __launch_bounds__(256) k_proj(
    const float* __restrict__ x, const float* __restrict__ Wq,
    const float* __restrict__ Wk, const float* __restrict__ Wv,
    __bf16* __restrict__ Q, __bf16* __restrict__ Kb, __bf16* __restrict__ Vb) {
  __shared__ __align__(16) __bf16 As[128 * 32];   // [m][k] row-major
  __shared__ __align__(16) __bf16 Bt[128 * 32];   // [n][k] (transposed tile)
  const int tid = threadIdx.x;
  const int lane = tid & 31, l16 = lane & 15, half = lane >> 4, w = tid >> 5;
  const int n0 = blockIdx.x * 128;
  const int m0 = blockIdx.y * 128;
  const int b = blockIdx.z / 3, sel = blockIdx.z % 3;
  const float* W = (sel == 0) ? Wq : (sel == 1) ? Wk : Wv;
  __bf16* O = (sel == 0) ? Q : (sel == 1) ? Kb : Vb;
  const float* xb = x + (size_t)b * CD * NN;

  v8f acc[8] = {};
  for (int kt = 0; kt < CD; kt += 32) {
    stage_a_f32(As, W + (size_t)m0 * CD + kt, CD, tid);
    stage_bt_f32(Bt, xb + (size_t)kt * NN + n0, NN, tid);
    // speculative WGP-scope prefetch of next K-step (global_prefetch_b8)
    __builtin_prefetch(&xb[(size_t)(kt + 32 + (tid >> 3)) * NN + n0 + (tid & 7) * 16], 0, 3);
    __builtin_prefetch(&W[(size_t)(m0 + (tid >> 1)) * CD + kt + 32 + (tid & 1) * 16], 0, 3);
    __syncthreads();
    const v16bf a = frag_a(&As[(w * 16 + l16) * 32], half);
#pragma unroll
    for (int nt = 0; nt < 8; ++nt) {
      const v16bf bb = frag_b(&Bt[(nt * 16 + l16) * 32], half);
      acc[nt] = wmma_bf16(a, bb, acc[nt]);
    }
    __syncthreads();
  }
#pragma unroll
  for (int nt = 0; nt < 8; ++nt)
#pragma unroll
    for (int j = 0; j < 8; ++j) {
      int m = m0 + w * 16 + j + 8 * half;
      O[((size_t)b * HD + m) * NN + n0 + nt * 16 + l16] = (__bf16)acc[nt][j];
    }
}

// ---- Kernel 2: K softmax over spatial axis, in-place (one row / block) -----
__global__ void __launch_bounds__(256) k_ksoftmax(__bf16* __restrict__ Kb) {
  __shared__ float red[256];
  const int tid = threadIdx.x;
  __bf16* row = Kb + (size_t)blockIdx.x * NN;
  float v[16];
#pragma unroll
  for (int i = 0; i < 16; ++i) v[i] = (float)row[tid + i * 256];
  float m = v[0];
#pragma unroll
  for (int i = 1; i < 16; ++i) m = fmaxf(m, v[i]);
  red[tid] = m; __syncthreads();
  for (int s = 128; s > 0; s >>= 1) {
    if (tid < s) red[tid] = fmaxf(red[tid], red[tid + s]);
    __syncthreads();
  }
  m = red[0]; __syncthreads();
  float sum = 0.f;
#pragma unroll
  for (int i = 0; i < 16; ++i) { v[i] = __expf(v[i] - m); sum += v[i]; }
  red[tid] = sum; __syncthreads();
  for (int s = 128; s > 0; s >>= 1) {
    if (tid < s) red[tid] += red[tid + s];
    __syncthreads();
  }
  const float inv = 1.f / red[0];
#pragma unroll
  for (int i = 0; i < 16; ++i) row[tid + i * 256] = (__bf16)(v[i] * inv);
}

// ---- Kernel 3: Q softmax over 64 head-channels (+*scale), in-place ---------
__global__ void __launch_bounds__(256) k_qsoftmax(__bf16* __restrict__ Q) {
  const int n = blockIdx.x * 256 + threadIdx.x;         // 0..4095
  const int bh = blockIdx.y;                            // 0..127
  const size_t base =
      ((size_t)(bh >> 3) * HD + (size_t)(bh & 7) * HC) * NN + n;
  float v[HC];
#pragma unroll
  for (int d = 0; d < HC; ++d) v[d] = (float)Q[base + (size_t)d * NN];
  float m = v[0];
#pragma unroll
  for (int d = 1; d < HC; ++d) m = fmaxf(m, v[d]);
  float s = 0.f;
#pragma unroll
  for (int d = 0; d < HC; ++d) { v[d] = __expf(v[d] - m); s += v[d]; }
  const float inv = 0.125f / s;  // scale = HC^-0.5 = 1/8
#pragma unroll
  for (int d = 0; d < HC; ++d)
    Q[base + (size_t)d * NN] = (__bf16)(v[d] * inv);
}

// ---- Kernel 4: context[bh] = ksm_h (64xN) @ v_h^T (Nx64), f32 out ----------
// Both operands are row-major [ch][n] with WMMA-K = n, so every fragment is
// two aligned global_load_b128 — no shuffles.
__global__ void __launch_bounds__(256) k_context(
    const __bf16* __restrict__ Kb, const __bf16* __restrict__ Vb,
    float* __restrict__ ctx) {
  const int tid = threadIdx.x;
  const int lane = tid & 31, l16 = lane & 15, half = lane >> 4, w = tid >> 5;
  const int bh = blockIdx.x;
  const __bf16* Kp = Kb + (size_t)bh * HC * NN;
  const __bf16* Vp = Vb + (size_t)bh * HC * NN;
  const int m_sub = w >> 1, nbase = (w & 1) * 2;
  v8f acc0 = {}, acc1 = {};
  for (int kt = 0; kt < NN; kt += 32) {
    const v16bf a = frag_a(Kp + (size_t)(m_sub * 16 + l16) * NN + kt, half);
    const v16bf b0 = frag_b(Vp + (size_t)((nbase + 0) * 16 + l16) * NN + kt, half);
    const v16bf b1 = frag_b(Vp + (size_t)((nbase + 1) * 16 + l16) * NN + kt, half);
    acc0 = wmma_bf16(a, b0, acc0);
    acc1 = wmma_bf16(a, b1, acc1);
  }
#pragma unroll
  for (int j = 0; j < 8; ++j) {
    const int m = m_sub * 16 + j + 8 * half;
    ctx[(size_t)bh * 4096 + m * 64 + (nbase + 0) * 16 + l16] = acc0[j];
    ctx[(size_t)bh * 4096 + m * 64 + (nbase + 1) * 16 + l16] = acc1[j];
  }
}

// ---- Kernel 5: Mbig[b][:, h*64+d] = Wo_h (256x64) @ ctx_h^T (64x64) --------
__global__ void __launch_bounds__(256) k_mbig(
    const float* __restrict__ Wo, const float* __restrict__ ctx,
    float* __restrict__ Mbig) {
  const int tid = threadIdx.x;
  const int lane = tid & 31, l16 = lane & 15, half = lane >> 4, w = tid >> 5;
  const int bh = blockIdx.x, b = bh >> 3, h = bh & 7;
  const float* Wop = Wo + h * HC;                 // row stride HD
  const float* cp = ctx + (size_t)bh * 4096;      // [d][e]
  for (int t = w * 8; t < w * 8 + 8; ++t) {
    const int m_sub = t >> 2, n_sub = t & 3;
    v8f acc = {};
#pragma unroll
    for (int kt = 0; kt < HC; kt += 32) {
      const float* pa = Wop + (size_t)(m_sub * 16 + l16) * HD + kt;
      const float* pb = cp + (size_t)(n_sub * 16 + l16) * 64 + kt;  // B[k][d]=ctx[d][k]
      v16bf a, bb;
#pragma unroll
      for (int i = 0; i < 8; ++i) {
        a[i]     = (__bf16)pa[8 * half + i];
        a[8 + i] = (__bf16)pa[16 + 8 * half + i];
        bb[i]     = (__bf16)pb[8 * half + i];
        bb[8 + i] = (__bf16)pb[16 + 8 * half + i];
      }
      acc = wmma_bf16(a, bb, acc);
    }
#pragma unroll
    for (int j = 0; j < 8; ++j) {
      const int o = m_sub * 16 + j + 8 * half;
      Mbig[((size_t)b * CD + o) * HD + h * HC + n_sub * 16 + l16] = acc[j];
    }
  }
}

// ---- Kernel 6: Y[b] = Mbig[b] (256x512) @ qsm[b] (512xN) + bo, f32 out -----
__global__ void __launch_bounds__(256) k_ygemm(
    const float* __restrict__ M, const __bf16* __restrict__ Q,
    const float* __restrict__ bo, float* __restrict__ Y) {
  __shared__ __align__(16) __bf16 As[128 * 32];
  __shared__ __align__(16) __bf16 Bt[128 * 32];
  const int tid = threadIdx.x;
  const int lane = tid & 31, l16 = lane & 15, half = lane >> 4, w = tid >> 5;
  const int n0 = blockIdx.x * 128;
  const int m0 = blockIdx.y * 128;
  const int b = blockIdx.z;
  const float* Ap = M + (size_t)b * CD * HD;
  const __bf16* Bp = Q + (size_t)b * HD * NN;

  v8f acc[8] = {};
  for (int kt = 0; kt < HD; kt += 32) {
    stage_a_f32(As, Ap + (size_t)m0 * HD + kt, HD, tid);
    stage_bt_bf16(Bt, Bp + (size_t)kt * NN + n0, NN, tid);
    __builtin_prefetch(&Bp[(size_t)(kt + 32 + (tid >> 3)) * NN + n0 + (tid & 7) * 16], 0, 3);
    __builtin_prefetch(&Ap[(size_t)(m0 + (tid >> 1)) * HD + kt + 32 + (tid & 1) * 16], 0, 3);
    __syncthreads();
    const v16bf a = frag_a(&As[(w * 16 + l16) * 32], half);
#pragma unroll
    for (int nt = 0; nt < 8; ++nt) {
      const v16bf bb = frag_b(&Bt[(nt * 16 + l16) * 32], half);
      acc[nt] = wmma_bf16(a, bb, acc[nt]);
    }
    __syncthreads();
  }
#pragma unroll
  for (int nt = 0; nt < 8; ++nt)
#pragma unroll
    for (int j = 0; j < 8; ++j) {
      const int m = m0 + w * 16 + j + 8 * half;
      Y[((size_t)b * CD + m) * NN + n0 + nt * 16 + l16] = acc[nt][j] + bo[m];
    }
}

// ---- Kernel 7/8: two-stage GroupNorm statistics ---------------------------
__global__ void __launch_bounds__(256) k_part(const float* __restrict__ Y,
                                              float* __restrict__ part) {
  __shared__ float s1[256], s2[256];
  const int tid = threadIdx.x;
  const float* p = Y + (size_t)blockIdx.x * 4096;
  float s = 0.f, q = 0.f;
#pragma unroll 4
  for (int i = tid; i < 4096; i += 256) { float v = p[i]; s += v; q += v * v; }
  s1[tid] = s; s2[tid] = q; __syncthreads();
  for (int st = 128; st > 0; st >>= 1) {
    if (tid < st) { s1[tid] += s1[tid + st]; s2[tid] += s2[tid + st]; }
    __syncthreads();
  }
  if (tid == 0) { part[blockIdx.x * 2] = s1[0]; part[blockIdx.x * 2 + 1] = s2[0]; }
}

__global__ void __launch_bounds__(256) k_stats(const float* __restrict__ part,
                                               float* __restrict__ stats) {
  __shared__ float s1[256], s2[256];
  const int tid = threadIdx.x, b = blockIdx.x;
  s1[tid] = part[(b * 256 + tid) * 2];
  s2[tid] = part[(b * 256 + tid) * 2 + 1];
  __syncthreads();
  for (int st = 128; st > 0; st >>= 1) {
    if (tid < st) { s1[tid] += s1[tid + st]; s2[tid] += s2[tid + st]; }
    __syncthreads();
  }
  if (tid == 0) {
    const float inv_n = 1.f / ((float)CD * (float)NN);
    float mean = s1[0] * inv_n;
    float var = s2[0] * inv_n - mean * mean;
    stats[b * 2] = mean;
    stats[b * 2 + 1] = rsqrtf(var + 1e-5f);
  }
}

// ---- Kernel 9: normalize + affine + residual ------------------------------
__global__ void __launch_bounds__(256) k_final(
    const float* __restrict__ Y, const float* __restrict__ x,
    const float* __restrict__ gamma, const float* __restrict__ beta,
    const float* __restrict__ stats, float* __restrict__ out) {
  const size_t e = ((size_t)blockIdx.x * 256 + threadIdx.x) * 4;
  const int b = (int)(e / ((size_t)CD * NN));
  const int c = (int)((e / NN) % CD);
  const float mean = stats[b * 2], inv = stats[b * 2 + 1];
  const float gg = gamma[c] * inv;
  const float off = beta[c] - mean * gg;
  const float4 yv = *(const float4*)(Y + e);
  const float4 xv = *(const float4*)(x + e);
  float4 o;
  o.x = yv.x * gg + off + xv.x;
  o.y = yv.y * gg + off + xv.y;
  o.z = yv.z * gg + off + xv.z;
  o.w = yv.w * gg + off + xv.w;
  *(float4*)(out + e) = o;
}

extern "C" void kernel_launch(void* const* d_in, const int* in_sizes, int n_in,
                              void* d_out, int out_size, void* d_ws,
                              size_t ws_size, hipStream_t stream) {
  (void)in_sizes; (void)n_in; (void)out_size; (void)ws_size;
  const float* x  = (const float*)d_in[0];
  const float* Wq = (const float*)d_in[1];
  const float* Wk = (const float*)d_in[2];
  const float* Wv = (const float*)d_in[3];
  const float* Wo = (const float*)d_in[4];
  const float* bo = (const float*)d_in[5];
  const float* gm = (const float*)d_in[6];
  const float* bt = (const float*)d_in[7];
  float* out = (float*)d_out;

  // Workspace layout (~266 MiB total)
  const size_t QKV_E = (size_t)BATCH * HD * NN;         // 33,554,432
  char* w = (char*)d_ws;
  __bf16* Q    = (__bf16*)w; w += QKV_E * 2;
  __bf16* Kb   = (__bf16*)w; w += QKV_E * 2;
  __bf16* Vb   = (__bf16*)w; w += QKV_E * 2;
  float* ctx   = (float*)w;  w += (size_t)BATCH * NHEAD * HC * HC * 4;
  float* Mbig  = (float*)w;  w += (size_t)BATCH * CD * HD * 4;
  float* Y     = (float*)w;  w += (size_t)BATCH * CD * NN * 4;
  float* part  = (float*)w;  w += (size_t)BATCH * 256 * 2 * 4;
  float* stats = (float*)w;

  k_proj<<<dim3(NN / 128, HD / 128, BATCH * 3), 256, 0, stream>>>(
      x, Wq, Wk, Wv, Q, Kb, Vb);
  k_ksoftmax<<<dim3(BATCH * HD), 256, 0, stream>>>(Kb);
  k_qsoftmax<<<dim3(NN / 256, BATCH * NHEAD), 256, 0, stream>>>(Q);
  k_context<<<dim3(BATCH * NHEAD), 256, 0, stream>>>(Kb, Vb, ctx);
  k_mbig<<<dim3(BATCH * NHEAD), 256, 0, stream>>>(Wo, ctx, Mbig);
  k_ygemm<<<dim3(NN / 128, CD / 128, BATCH), 256, 0, stream>>>(Mbig, Q, bo, Y);
  k_part<<<dim3(BATCH * 256), 256, 0, stream>>>(Y, part);
  k_stats<<<dim3(BATCH), 256, 0, stream>>>(part, stats);
  k_final<<<dim3((BATCH * CD * NN) / (256 * 4)), 256, 0, stream>>>(
      Y, x, gm, bt, stats, out);
}
